// TrustGNN_55422257987980
// MI455X (gfx1250) — compile-verified, matrix operand order checked by reading, not run on hardware
//
#include <hip/hip_runtime.h>
#include <hip/hip_bf16.h>
#include <stdint.h>

// ---------------------------------------------------------------------------
// TrustGNN forward, CDNA5 (gfx1250) wave32 + bf16 WMMA implementation.
//   edge MLP fused per 16-edge tile (no [E,770] materialization)
//   all GEMMs via v_wmma_f32_16x16x32_bf16, fp32 accumulate
//   A-tile staging via GLOBAL_LOAD_ASYNC_TO_LDS_B128 when available
//   scatter-add via global_atomic_add_f32
// ---------------------------------------------------------------------------

typedef __attribute__((ext_vector_type(16))) __bf16 v16bf;
typedef __attribute__((ext_vector_type(8)))  float  v8f;

#define EMB   384
#define K1P   416   // 385 padded to 13*32
#define N1P   832   // 770 padded to 13*64
#define K2P   832   // 770 padded to 26*32
#define N1RAW 770

// ---- async copy to LDS (CDNA5) -------------------------------------------
#if defined(__HIP_DEVICE_COMPILE__) && \
    __has_builtin(__builtin_amdgcn_global_load_async_to_lds_b128)
#define HAVE_ASYNC_COPY 1
#else
#define HAVE_ASYNC_COPY 0
#endif

#if HAVE_ASYNC_COPY
// builtin signature (from hipcc diagnostic): param0 is a global-AS pointer to
// a 16-byte int vector; param1 the LDS-side pointer; then imm offset, imm cpol.
typedef int vi4 __attribute__((ext_vector_type(4)));
typedef __attribute__((address_space(1))) vi4* gvi4p;
typedef __attribute__((address_space(3))) vi4* lvi4p;

__device__ __forceinline__ void async_wait0() {
#if __has_builtin(__builtin_amdgcn_s_wait_asynccnt)
    __builtin_amdgcn_s_wait_asynccnt(0);
#else
    asm volatile("s_wait_asynccnt 0x0" ::: "memory");
#endif
}
#endif

// Copy nchunks * 16B from global to LDS, wave-cooperative.
__device__ __forceinline__ void stage16(const uint16_t* __restrict__ gsrc,
                                        uint16_t* ldst, int nchunks, int lane) {
#if HAVE_ASYNC_COPY
    const __attribute__((address_space(1))) char* g =
        (const __attribute__((address_space(1))) char*)gsrc;
    __attribute__((address_space(3))) char* l =
        (__attribute__((address_space(3))) char*)ldst;
    for (int i = lane; i < nchunks; i += 32)
        __builtin_amdgcn_global_load_async_to_lds_b128(
            (gvi4p)(g + (size_t)i * 16), (lvi4p)(l + (size_t)i * 16), 0, 0);
    async_wait0();
#else
    const uint4* s = (const uint4*)gsrc;
    uint4* d = (uint4*)ldst;
    for (int i = lane; i < nchunks; i += 32) d[i] = s[i];
#endif
}

// Per-lane copy of nchunks * 16B (each lane copies its own row segment).
__device__ __forceinline__ void stage16_lane(const uint16_t* __restrict__ gsrc,
                                             uint16_t* ldst, int nchunks) {
#if HAVE_ASYNC_COPY
    const __attribute__((address_space(1))) char* g =
        (const __attribute__((address_space(1))) char*)gsrc;
    __attribute__((address_space(3))) char* l =
        (__attribute__((address_space(3))) char*)ldst;
    for (int i = 0; i < nchunks; ++i)
        __builtin_amdgcn_global_load_async_to_lds_b128(
            (gvi4p)(g + (size_t)i * 16), (lvi4p)(l + (size_t)i * 16), 0, 0);
#else
    const uint4* s = (const uint4*)gsrc;
    uint4* d = (uint4*)ldst;
    for (int i = 0; i < nchunks; ++i) d[i] = s[i];
#endif
}

__device__ __forceinline__ void stage_lane_fence() {
#if HAVE_ASYNC_COPY
    async_wait0();
#endif
}

// ---- scalar helpers --------------------------------------------------------
__device__ __forceinline__ uint16_t f2bf(float f) {
    uint32_t u = __float_as_uint(f);
    u += 0x7FFFu + ((u >> 16) & 1u);        // round-to-nearest-even
    return (uint16_t)(u >> 16);
}
__device__ __forceinline__ float bf2f(uint16_t h) {
    return __uint_as_float(((uint32_t)h) << 16);
}

// Load one 16-lane-striped bf16 fragment: two contiguous 16B runs,
// K offsets [0,8) and [16,24) relative to p (caller folds in 8*half).
__device__ __forceinline__ v16bf ldfrag(const uint16_t* p) {
    union { uint4 q[2]; v16bf v; } u;
    u.q[0] = *(const uint4*)(p);
    u.q[1] = *(const uint4*)(p + 16);
    return u.v;
}

__device__ __forceinline__ v8f wmma_bf16(v16bf a, v16bf b, v8f c) {
    return __builtin_amdgcn_wmma_f32_16x16x32_bf16(
        false, a, false, b, (short)0, c, false, false);
}

__device__ __forceinline__ v8f vzero8() {
    v8f z = {0.f, 0.f, 0.f, 0.f, 0.f, 0.f, 0.f, 0.f};
    return z;
}

// ---------------------------------------------------------------------------
// Packing / utility kernels
// ---------------------------------------------------------------------------

// W [K][Nn] fp32 row-major  ->  Wt [Npad][Kpad] bf16 (transposed, zero padded)
__global__ void pack_wt_kernel(const float* __restrict__ W, uint16_t* __restrict__ Wt,
                               int K, int Nn, int Kpad, int Npad) {
    long total = (long)Kpad * Npad;
    for (long i = (long)blockIdx.x * blockDim.x + threadIdx.x; i < total;
         i += (long)gridDim.x * blockDim.x) {
        int n = (int)(i / Kpad);
        int k = (int)(i % Kpad);
        float v = (k < K && n < Nn) ? W[(size_t)k * Nn + n] : 0.f;
        Wt[i] = f2bf(v);
    }
}

// edge_attr = concat([trust, qemb]) -> bf16 [E][K1P], cols 385..415 zero
__global__ void pack_eattr_kernel(const float* __restrict__ trust,
                                  const float* __restrict__ qemb,
                                  uint16_t* __restrict__ eattr, long E) {
    long total = E * K1P;
    for (long i = (long)blockIdx.x * blockDim.x + threadIdx.x; i < total;
         i += (long)gridDim.x * blockDim.x) {
        long e = i / K1P;
        int  c = (int)(i % K1P);
        float v = (c == 0) ? trust[e]
                : (c <= EMB) ? qemb[e * EMB + (c - 1)] : 0.f;
        eattr[i] = f2bf(v);
    }
}

__global__ void cvt_bf16_kernel(const float* __restrict__ s, uint16_t* __restrict__ d, long n) {
    for (long i = (long)blockIdx.x * blockDim.x + threadIdx.x; i < n;
         i += (long)gridDim.x * blockDim.x)
        d[i] = f2bf(s[i]);
}

__global__ void zerof_kernel(float* __restrict__ p, long n) {
    for (long i = (long)blockIdx.x * blockDim.x + threadIdx.x; i < n;
         i += (long)gridDim.x * blockDim.x)
        p[i] = 0.f;
}

// ---------------------------------------------------------------------------
// Fused edge-layer kernel: one wave per 16-edge tile.
//   h    = relu(eattr_tile @ W1t^T + b1)   (16 x 770, kept bf16 in LDS)
//   et   = h @ W2t^T + b2                  (16 x N2)
//   aggr[col[e]] += et[m] + xin[row[e]]    (atomic scatter-add)
// ---------------------------------------------------------------------------
__global__ void __launch_bounds__(32)
edge_layer_kernel(const uint16_t* __restrict__ eattr,  // [E][K1P]
                  const uint16_t* __restrict__ W1t,    // [N1P][K1P]
                  const float*    __restrict__ b1,     // [770]
                  const uint16_t* __restrict__ W2t,    // [N2][K2P]
                  const float*    __restrict__ b2,     // [N2]
                  const float*    __restrict__ xin,    // [N][N2]
                  const int*      __restrict__ row,
                  const int*      __restrict__ col,
                  float*          __restrict__ aggr,   // [N][N2]
                  int N2) {
    __shared__ __align__(16) uint16_t atile[16 * K1P];  // 13.0 KB
    __shared__ __align__(16) uint16_t hbuf [16 * K2P];  // 26.0 KB

    const int lane = threadIdx.x & 31;
    const int half = lane >> 4;
    const int nlo  = lane & 15;
    const int koff = half * 8;
    const long e0  = (long)blockIdx.x * 16;

    // stage edge-attr tile (contiguous 13.3 KB) into LDS (async if available)
    stage16(eattr + e0 * K1P, atile, 16 * K1P / 8, lane);
    __syncthreads();

    // ---- stage A: 16x832 = relu(A(16x416) x W1t) ----
    for (int nb = 0; nb < N1P / 64; ++nb) {
        v8f acc[4];
#pragma unroll
        for (int j = 0; j < 4; ++j) acc[j] = vzero8();
        const int nbase = nb * 64;
#pragma unroll
        for (int k = 0; k < K1P / 32; ++k) {
            const int kb = k * 32 + koff;
            v16bf a = ldfrag(atile + nlo * K1P + kb);
#pragma unroll
            for (int j = 0; j < 4; ++j) {
                v16bf b = ldfrag(W1t + (size_t)(nbase + j * 16 + nlo) * K1P + kb);
                acc[j] = wmma_bf16(a, b, acc[j]);
            }
        }
#pragma unroll
        for (int j = 0; j < 4; ++j) {
            const int n = nbase + j * 16 + nlo;
            const float bias = (n < N1RAW) ? b1[n] : 0.f;
#pragma unroll
            for (int g = 0; g < 8; ++g) {
                float v = acc[j][g] + bias;
                v = v > 0.f ? v : 0.f;
                hbuf[(size_t)(g + 8 * half) * K2P + n] = f2bf(v);
            }
        }
    }
    __syncthreads();

    int re[8], ce[8];
#pragma unroll
    for (int g = 0; g < 8; ++g) {
        const long e = e0 + g + 8 * half;
        re[g] = row[e];
        ce[g] = col[e];
    }

    // ---- stage B: et(16xN2) = h(16x832) x W2t ; gather-add + scatter ----
    const int nb2 = N2 >> 6;
    for (int nb = 0; nb < nb2; ++nb) {
        v8f acc[4];
#pragma unroll
        for (int j = 0; j < 4; ++j) acc[j] = vzero8();
        const int nbase = nb * 64;
        for (int k = 0; k < K2P / 32; ++k) {
            const int kb = k * 32 + koff;
            v16bf a = ldfrag(hbuf + nlo * K2P + kb);
#pragma unroll
            for (int j = 0; j < 4; ++j) {
                v16bf b = ldfrag(W2t + (size_t)(nbase + j * 16 + nlo) * K2P + kb);
                acc[j] = wmma_bf16(a, b, acc[j]);
            }
        }
#pragma unroll
        for (int j = 0; j < 4; ++j) {
            const int n = nbase + j * 16 + nlo;
            const float bias = b2[n];
#pragma unroll
            for (int g = 0; g < 8; ++g) {
                float v = acc[j][g] + bias + xin[(size_t)re[g] * N2 + n];
                atomicAdd(&aggr[(size_t)ce[g] * N2 + n], v);
            }
        }
    }
}

// ---------------------------------------------------------------------------
// Node linear: out = relu(A(bf16) @ Wt^T + bias); one wave per 16-row tile
// ---------------------------------------------------------------------------
__global__ void __launch_bounds__(32)
node_layer_kernel(const uint16_t* __restrict__ A,    // [N][K] bf16
                  const uint16_t* __restrict__ Wt,   // [Nout][K]
                  const float*    __restrict__ bias, // [Nout]
                  float*          __restrict__ out,  // [N][Nout]
                  int K, int Nout) {
    __shared__ __align__(16) uint16_t atile[16 * EMB];  // max K = 384

    const int lane = threadIdx.x & 31;
    const int half = lane >> 4;
    const int nlo  = lane & 15;
    const int koff = half * 8;
    const long r0  = (long)blockIdx.x * 16;

    stage16(A + r0 * K, atile, 16 * K / 8, lane);
    __syncthreads();

    for (int nb = 0; nb < (Nout >> 6); ++nb) {
        v8f acc[4];
#pragma unroll
        for (int j = 0; j < 4; ++j) acc[j] = vzero8();
        const int nbase = nb * 64;
        for (int k = 0; k < (K >> 5); ++k) {
            const int kb = k * 32 + koff;
            v16bf a = ldfrag(atile + nlo * K + kb);
#pragma unroll
            for (int j = 0; j < 4; ++j) {
                v16bf b = ldfrag(Wt + (size_t)(nbase + j * 16 + nlo) * K + kb);
                acc[j] = wmma_bf16(a, b, acc[j]);
            }
        }
#pragma unroll
        for (int j = 0; j < 4; ++j) {
            const int n = nbase + j * 16 + nlo;
            const float bs = bias[n];
#pragma unroll
            for (int g = 0; g < 8; ++g) {
                float v = acc[j][g] + bs;
                out[(size_t)(r0 + g + 8 * half) * Nout + n] = v > 0.f ? v : 0.f;
            }
        }
    }
}

// ---------------------------------------------------------------------------
// Predictor: per 16-edge tile
//   pred_in = [h2[row] | h2[col] | qemb]  (16 x 640, bf16 in LDS)
//   p = relu(pred_in @ Wp1t^T + bp1)      (16 x 256, bf16 in LDS)
//   out = sigmoid(p . Wp2 + bp2)
// ---------------------------------------------------------------------------
__global__ void __launch_bounds__(32)
predictor_kernel(const uint16_t* __restrict__ h2b,   // [N][128] bf16
                 const float*    __restrict__ qemb,  // [E][384]
                 const int*      __restrict__ row,
                 const int*      __restrict__ col,
                 const uint16_t* __restrict__ Wp1t,  // [256][640]
                 const float*    __restrict__ bp1,   // [256]
                 const float*    __restrict__ Wp2,   // [256]
                 const float*    __restrict__ bp2,   // [1]
                 float*          __restrict__ out) { // [E]
    const int KP = 640, NP = 256;
    __shared__ __align__(16) uint16_t atile[16 * 640]; // 20 KB
    __shared__ __align__(16) uint16_t pbuf [16 * 256]; //  8 KB

    const int lane = threadIdx.x & 31;
    const int half = lane >> 4;
    const int nlo  = lane & 15;
    const int koff = half * 8;
    const long e0  = (long)blockIdx.x * 16;

    // build pred_in tile: 2 lanes per row (m = lane&15, s = lane>>4)
    {
        const int m = lane & 15, s = lane >> 4;
        const long e = e0 + m;
        const int rr = row[e], cc = col[e];
        // h2 segment: lane copies its row's 128 bf16 (16 x 16B) async
        stage16_lane(h2b + (size_t)(s ? cc : rr) * 128,
                     atile + m * KP + s * 128, 16);
        // qemb segment: fp32 -> bf16 conversion (192 cols per lane)
        for (int c = s * 192; c < s * 192 + 192; ++c)
            atile[m * KP + 256 + c] = f2bf(qemb[(size_t)e * EMB + c]);
        stage_lane_fence();
    }
    __syncthreads();

    for (int nb = 0; nb < NP / 64; ++nb) {
        v8f acc[4];
#pragma unroll
        for (int j = 0; j < 4; ++j) acc[j] = vzero8();
        const int nbase = nb * 64;
#pragma unroll
        for (int k = 0; k < KP / 32; ++k) {
            const int kb = k * 32 + koff;
            v16bf a = ldfrag(atile + nlo * KP + kb);
#pragma unroll
            for (int j = 0; j < 4; ++j) {
                v16bf b = ldfrag(Wp1t + (size_t)(nbase + j * 16 + nlo) * KP + kb);
                acc[j] = wmma_bf16(a, b, acc[j]);
            }
        }
#pragma unroll
        for (int j = 0; j < 4; ++j) {
            const int n = nbase + j * 16 + nlo;
            const float bs = bp1[n];
#pragma unroll
            for (int g = 0; g < 8; ++g) {
                float v = acc[j][g] + bs;
                v = v > 0.f ? v : 0.f;
                pbuf[(g + 8 * half) * NP + n] = f2bf(v);
            }
        }
    }
    __syncthreads();

    // final 256->1 dot + sigmoid: 2 lanes per row, halves of K
    {
        const int m = lane & 15, s = lane >> 4;
        float sum = 0.f;
        for (int k = s * 128; k < s * 128 + 128; ++k)
            sum += bf2f(pbuf[m * NP + k]) * Wp2[k];
        float tot = sum + __shfl_xor(sum, 16, 32);
        if (s == 0) {
            float z = tot + bp2[0];
            out[e0 + m] = 1.f / (1.f + expf(-z));
        }
    }
}

// ---------------------------------------------------------------------------
// Host-side launch
// ---------------------------------------------------------------------------
extern "C" void kernel_launch(void* const* d_in, const int* in_sizes, int n_in,
                              void* d_out, int out_size, void* d_ws, size_t ws_size,
                              hipStream_t stream) {
    const float* x      = (const float*)d_in[0];
    const int*   eidx   = (const int*)  d_in[1];
    const float* trust  = (const float*)d_in[2];
    const float* qemb   = (const float*)d_in[3];
    const float* W_e1a  = (const float*)d_in[4];
    const float* b_e1a  = (const float*)d_in[5];
    const float* W_e1b  = (const float*)d_in[6];
    const float* b_e1b  = (const float*)d_in[7];
    const float* W_n1   = (const float*)d_in[8];
    const float* b_n1   = (const float*)d_in[9];
    const float* W_e2a  = (const float*)d_in[10];
    const float* b_e2a  = (const float*)d_in[11];
    const float* W_e2b  = (const float*)d_in[12];
    const float* b_e2b  = (const float*)d_in[13];
    const float* W_n2   = (const float*)d_in[14];
    const float* b_n2   = (const float*)d_in[15];
    const float* W_p1   = (const float*)d_in[16];
    const float* b_p1   = (const float*)d_in[17];
    const float* W_p2   = (const float*)d_in[18];
    const float* b_p2   = (const float*)d_in[19];

    const long E  = in_sizes[2];            // 400000
    const long NN = in_sizes[0] / EMB;      // 50000
    const int  HID = 256, OUT = 128, PIN = 640;
    const int*   rowp = eidx;
    const int*   colp = eidx + E;
    float*       outp = (float*)d_out;

    // workspace carve-up (256B aligned)
    char* base = (char*)d_ws;
    size_t off = 0;
    auto alloc = [&](size_t bytes) -> char* {
        char* p = base + off;
        off = (off + bytes + 255) & ~(size_t)255;
        return p;
    };
    uint16_t* eattr  = (uint16_t*)alloc((size_t)E * K1P * 2);
    uint16_t* W1t_1  = (uint16_t*)alloc((size_t)N1P * K1P * 2);
    uint16_t* W2t_1  = (uint16_t*)alloc((size_t)EMB * K2P * 2);
    uint16_t* Wn1t   = (uint16_t*)alloc((size_t)HID * EMB * 2);
    uint16_t* W1t_2  = (uint16_t*)alloc((size_t)N1P * K1P * 2);
    uint16_t* W2t_2  = (uint16_t*)alloc((size_t)HID * K2P * 2);
    uint16_t* Wn2t   = (uint16_t*)alloc((size_t)OUT * HID * 2);
    uint16_t* Wp1t   = (uint16_t*)alloc((size_t)HID * PIN * 2);
    float*    aggr1  = (float*)   alloc((size_t)NN * EMB * 4);
    uint16_t* aggr1b = (uint16_t*)alloc((size_t)NN * EMB * 2);
    float*    h1     = (float*)   alloc((size_t)NN * HID * 4);
    float*    aggr2  = (float*)   alloc((size_t)NN * HID * 4);
    uint16_t* aggr2b = (uint16_t*)alloc((size_t)NN * HID * 2);
    float*    h2     = (float*)   alloc((size_t)NN * OUT * 4);
    uint16_t* h2b    = (uint16_t*)alloc((size_t)NN * OUT * 2);

    auto gs = [](long n) { return (unsigned)((n + 255) / 256); };

    // ---- packing ----
    pack_eattr_kernel<<<gs(E * K1P), 256, 0, stream>>>(trust, qemb, eattr, E);
    pack_wt_kernel<<<gs((long)N1P * K1P), 256, 0, stream>>>(W_e1a, W1t_1, 385, 770, K1P, N1P);
    pack_wt_kernel<<<gs((long)EMB * K2P), 256, 0, stream>>>(W_e1b, W2t_1, 770, EMB, K2P, EMB);
    pack_wt_kernel<<<gs((long)HID * EMB), 256, 0, stream>>>(W_n1, Wn1t, EMB, HID, EMB, HID);
    pack_wt_kernel<<<gs((long)N1P * K1P), 256, 0, stream>>>(W_e2a, W1t_2, 385, 770, K1P, N1P);
    pack_wt_kernel<<<gs((long)HID * K2P), 256, 0, stream>>>(W_e2b, W2t_2, 770, HID, K2P, HID);
    pack_wt_kernel<<<gs((long)OUT * HID), 256, 0, stream>>>(W_n2, Wn2t, HID, OUT, HID, OUT);
    pack_wt_kernel<<<gs((long)HID * PIN), 256, 0, stream>>>(W_p1, Wp1t, PIN, HID, PIN, HID);

    const unsigned etiles = (unsigned)(E / 16);   // 25000
    const unsigned ntiles = (unsigned)(NN / 16);  // 3125

    // ---- layer 1 ----
    zerof_kernel<<<gs((long)NN * EMB), 256, 0, stream>>>(aggr1, (long)NN * EMB);
    edge_layer_kernel<<<etiles, 32, 0, stream>>>(eattr, W1t_1, b_e1a, W2t_1, b_e1b,
                                                 x, rowp, colp, aggr1, EMB);
    cvt_bf16_kernel<<<gs((long)NN * EMB), 256, 0, stream>>>(aggr1, aggr1b, (long)NN * EMB);
    node_layer_kernel<<<ntiles, 32, 0, stream>>>(aggr1b, Wn1t, b_n1, h1, EMB, HID);

    // ---- layer 2 ----
    zerof_kernel<<<gs((long)NN * HID), 256, 0, stream>>>(aggr2, (long)NN * HID);
    edge_layer_kernel<<<etiles, 32, 0, stream>>>(eattr, W1t_2, b_e2a, W2t_2, b_e2b,
                                                 h1, rowp, colp, aggr2, HID);
    cvt_bf16_kernel<<<gs((long)NN * HID), 256, 0, stream>>>(aggr2, aggr2b, (long)NN * HID);
    node_layer_kernel<<<ntiles, 32, 0, stream>>>(aggr2b, Wn2t, b_n2, h2, HID, OUT);
    cvt_bf16_kernel<<<gs((long)NN * OUT), 256, 0, stream>>>(h2, h2b, (long)NN * OUT);

    // ---- predictor ----
    predictor_kernel<<<etiles, 32, 0, stream>>>(h2b, qemb, rowp, colp,
                                                Wp1t, b_p1, W_p2, b_p2, outp);
}
